// MultiHeadedAttention_21268678049940
// MI455X (gfx1250) — compile-verified, matrix operand order checked by reading, not run on hardware
//
#include <hip/hip_runtime.h>
#include <hip/hip_bf16.h>

#define NUM_HEAD 12
#define DMODEL   768
#define HEAD_DIM 64
#define SEQ      2048
#define BATCH    2

typedef __attribute__((ext_vector_type(16))) __bf16 v16bf;
typedef __attribute__((ext_vector_type(8)))  __bf16 v8bf;
typedef __attribute__((ext_vector_type(8)))  float  v8f;

// Load a 16x32 bf16 WMMA A-fragment (or B-fragment from an n-major, i.e.
// transposed, matrix) from row-major storage with leading dim `ld` elements.
// ISA 16-bit A layout: lane L holds row m=L&15; half-select hs=L>>4;
// halves 0..7 -> K = k0 + hs*8 + j, halves 8..15 -> K = k0 + 16 + hs*8 + j.
__device__ __forceinline__ v16bf load_frag16(const __bf16* __restrict__ base,
                                             int ld, int row0, int k0) {
  const int lane = threadIdx.x & 31;
  const int m  = lane & 15;
  const int hs = lane >> 4;
  const __bf16* p = base + (size_t)(row0 + m) * ld + k0 + hs * 8;
  v8bf lo = *(const v8bf*)(p);        // 16B load
  v8bf hi = *(const v8bf*)(p + 16);   // 16B load
  v16bf f;
#pragma unroll
  for (int j = 0; j < 8; ++j) { f[j] = lo[j]; f[j + 8] = hi[j]; }
  return f;
}

// gfx1250 async global->LDS copy, 16B per lane, tracked with ASYNCcnt.
__device__ __forceinline__ void async_copy_b128(unsigned lds_byte_addr,
                                                const void* gaddr) {
  asm volatile("global_load_async_to_lds_b128 %0, %1, off"
               :: "v"(lds_byte_addr), "v"(gaddr) : "memory");
}

__global__ void cast_bf16_kernel(const float* __restrict__ in,
                                 __bf16* __restrict__ out, int n) {
  int i = blockIdx.x * blockDim.x + threadIdx.x;
  if (i < n) out[i] = (__bf16)in[i];
}

// out[n][k] = (bf16) in[k][n]  (DMODEL x DMODEL)
__global__ void transpose_cast_kernel(const float* __restrict__ in,
                                      __bf16* __restrict__ out) {
  int i = blockIdx.x * blockDim.x + threadIdx.x;
  if (i >= DMODEL * DMODEL) return;
  int n = i / DMODEL, k = i % DMODEL;
  out[i] = (__bf16)in[(size_t)k * DMODEL + n];
}

// One wave computes a 16x64 tile of one projection. Copy-free double-buffered
// K-loop (unrolled x2): each buffer is consumed by 4 WMMAs then immediately
// refilled with the k+2 fragments -- no register rotation movs.
__global__ __launch_bounds__(256)
void qkv_gemm_kernel(const __bf16* __restrict__ X,
                     const __bf16* __restrict__ WqT,
                     const __bf16* __restrict__ WkT,
                     const __bf16* __restrict__ WvT,
                     const float* __restrict__ bq,
                     const float* __restrict__ bk,
                     const float* __restrict__ bv,
                     __bf16* __restrict__ Qh,   // [b,h,s,hd]
                     __bf16* __restrict__ Kh,   // [b,h,s,hd]
                     __bf16* __restrict__ Vt) { // [b,h,hd,s]
  const int wave = threadIdx.x >> 5;
  const int lane = threadIdx.x & 31;
  const int nl  = lane & 15;
  const int hsv = lane >> 4;

  const int MT = (BATCH * SEQ) / 16; // 256
  const int NT = DMODEL / 64;        // 12
  int job  = blockIdx.x * (blockDim.x >> 5) + wave;
  int proj = job / (MT * NT);
  int rem  = job % (MT * NT);
  int mt   = rem / NT;
  int n64  = rem % NT;
  if (proj >= 3) return;

  const __bf16* WT  = (proj == 0) ? WqT : (proj == 1 ? WkT : WvT);
  const float* bias = (proj == 0) ? bq  : (proj == 1 ? bk  : bv);

  const int KSTEPS = DMODEL / 32; // 24 (even)
  v8f acc[4] = {};

  // Prime both buffers: A-set = step 0, B-set = step 1.
  v16bf aA = load_frag16(X, DMODEL, mt * 16, 0);
  v16bf bA[4];
#pragma unroll
  for (int nt = 0; nt < 4; ++nt)
    bA[nt] = load_frag16(WT, DMODEL, n64 * 64 + nt * 16, 0);
  v16bf aB = load_frag16(X, DMODEL, mt * 16, 32);
  v16bf bB[4];
#pragma unroll
  for (int nt = 0; nt < 4; ++nt)
    bB[nt] = load_frag16(WT, DMODEL, n64 * 64 + nt * 16, 32);

#pragma unroll 1
  for (int kk = 0; kk < KSTEPS; kk += 2) {
    // compute step kk from A-set
#pragma unroll
    for (int nt = 0; nt < 4; ++nt)
      acc[nt] = __builtin_amdgcn_wmma_f32_16x16x32_bf16(
          false, aA, false, bA[nt], (short)0, acc[nt], false, false);
    // refill A-set with step kk+2 (clamped dummy on the tail)
    {
      const int kp = (kk + 2 < KSTEPS) ? (kk + 2) : 0;
      aA = load_frag16(X, DMODEL, mt * 16, kp * 32);
#pragma unroll
      for (int nt = 0; nt < 4; ++nt)
        bA[nt] = load_frag16(WT, DMODEL, n64 * 64 + nt * 16, kp * 32);
    }
    // compute step kk+1 from B-set
#pragma unroll
    for (int nt = 0; nt < 4; ++nt)
      acc[nt] = __builtin_amdgcn_wmma_f32_16x16x32_bf16(
          false, aB, false, bB[nt], (short)0, acc[nt], false, false);
    // refill B-set with step kk+3
    {
      const int kp = (kk + 3 < KSTEPS) ? (kk + 3) : 0;
      aB = load_frag16(X, DMODEL, mt * 16, kp * 32);
#pragma unroll
      for (int nt = 0; nt < 4; ++nt)
        bB[nt] = load_frag16(WT, DMODEL, n64 * 64 + nt * 16, kp * 32);
    }
  }

#pragma unroll
  for (int nt = 0; nt < 4; ++nt) {
    int n = n64 * 64 + nt * 16 + nl;
    float bval = bias[n];
    int h = n >> 6, d = n & 63;
#pragma unroll
    for (int r = 0; r < 8; ++r) {
      int m = mt * 16 + r + 8 * hsv;
      int b = m >> 11, s = m & (SEQ - 1);
      float val = acc[nt][r] + bval;
      if (proj == 2) {
        Vt[(((size_t)(b * NUM_HEAD + h) * HEAD_DIM + d) * SEQ) + s] = (__bf16)val;
      } else {
        __bf16* dst = (proj == 0) ? Qh : Kh;
        dst[(((size_t)(b * NUM_HEAD + h) * SEQ + s) * HEAD_DIM) + d] = (__bf16)val;
      }
    }
  }
}

// Flash attention. One block = one (b,h) head-batch, 8 waves = 8 consecutive
// 16-query tiles. K/V tiles (64 keys) are streamed block-cooperatively into
// LDS with double-buffered async global->LDS copies (ASYNCcnt), so every
// wave reuses the shared tiles and the DMA overlaps WMMA + softmax.
__global__ __launch_bounds__(256)
void attn_kernel(const __bf16* __restrict__ Qh,
                 const __bf16* __restrict__ Kh,
                 const __bf16* __restrict__ Vg,
                 const float* __restrict__ mask, // [B,1,1,S]
                 float* __restrict__ out) {      // [B,S,D] f32
  __shared__ __align__(16) __bf16 Ks[2][64 * 64];   // [key][d]
  __shared__ __align__(16) __bf16 Vs[2][64 * 64];   // [d][key]
  __shared__ __align__(16) __bf16 Plds[8][16 * 64]; // per-wave P staging
  const int wave = threadIdx.x >> 5;
  const int lane = threadIdx.x & 31;
  const int nl  = lane & 15;
  const int hsv = lane >> 4;

  const int bh = blockIdx.x >> 4;              // 0..23
  const int qt = (blockIdx.x & 15) * 8 + wave; // 0..127
  const int b = bh / NUM_HEAD;
  const int h = bh % NUM_HEAD;

  const __bf16* Q = Qh + (size_t)bh * SEQ * HEAD_DIM;
  const __bf16* K = Kh + (size_t)bh * SEQ * HEAD_DIM;
  const __bf16* V = Vg + (size_t)bh * HEAD_DIM * SEQ;
  const float* mrow = mask + (size_t)b * SEQ;

  v16bf qfrag[2];
  qfrag[0] = load_frag16(Q, HEAD_DIM, qt * 16, 0);
  qfrag[1] = load_frag16(Q, HEAD_DIM, qt * 16, 32);

  v8f oacc[4] = {};
  float rmax[8], rsum[8];
#pragma unroll
  for (int r = 0; r < 8; ++r) { rmax[r] = -1e30f; rsum[r] = 0.f; }

  const float scale = 0.125f; // 1/sqrt(64)
  __bf16* myP = &Plds[wave][0];

  // Issue async tile loads for chunk `ci` into buffer `buf` (4 instrs/wave).
  auto issue_tiles = [&](int ci, int buf) {
    const int kb = ci * 64;
    const char* kg = (const char*)(K + (size_t)kb * HEAD_DIM); // 8KB contig
    const char* vg = (const char*)V;                            // [d][s]
    unsigned kl = (unsigned)(uintptr_t)(&Ks[buf][0]);
    unsigned vl = (unsigned)(uintptr_t)(&Vs[buf][0]);
#pragma unroll
    for (int j = 0; j < 2; ++j) {
      int c = (int)threadIdx.x + j * 256;       // 0..511, 16B chunks
      async_copy_b128(kl + c * 16, kg + (size_t)c * 16);
      int row = c >> 3;                         // d
      int col = (c & 7) * 16;                   // byte offset in 128B row
      async_copy_b128(vl + row * 128 + col,
                      vg + (size_t)row * (SEQ * 2) + (size_t)kb * 2 + col);
    }
  };

  issue_tiles(0, 0);

  const int NCH = SEQ / 64; // 32
  for (int ci = 0; ci < NCH; ++ci) {
    const int cur = ci & 1;
    const bool have_next = (ci + 1) < NCH;
    if (have_next) issue_tiles(ci + 1, cur ^ 1);
    if (have_next) asm volatile("s_wait_asynccnt 0x4" ::: "memory");
    else           asm volatile("s_wait_asynccnt 0x0" ::: "memory");
    __syncthreads(); // all waves' portions of buf[cur] visible

    const int kb = ci * 64;
    const __bf16* Kt = &Ks[cur][0];
    const __bf16* Vt = &Vs[cur][0];

    // scores: S = Q @ K^T (K tile in LDS, [key][d], ld=64)
    v8f sacc[4] = {};
#pragma unroll
    for (int ct = 0; ct < 4; ++ct) {
#pragma unroll
      for (int ks = 0; ks < 2; ++ks) {
        v16bf kf = load_frag16(Kt, 64, ct * 16, ks * 32);
        sacc[ct] = __builtin_amdgcn_wmma_f32_16x16x32_bf16(
            false, qfrag[ks], false, kf, (short)0, sacc[ct], false, false);
      }
    }
    float maskv[4];
#pragma unroll
    for (int ct = 0; ct < 4; ++ct) maskv[ct] = mrow[kb + ct * 16 + nl];

    // online softmax (fp32). C layout: row m = r + 8*hsv, col n = nl.
#pragma unroll
    for (int r = 0; r < 8; ++r) {
      float mx = rmax[r];
#pragma unroll
      for (int ct = 0; ct < 4; ++ct) {
        float sv = sacc[ct][r] * scale + maskv[ct];
        sacc[ct][r] = sv;
        mx = fmaxf(mx, sv);
      }
#pragma unroll
      for (int off = 8; off >= 1; off >>= 1)
        mx = fmaxf(mx, __shfl_xor(mx, off, 16));
      float corr = __expf(rmax[r] - mx);
      rmax[r] = mx;
      rsum[r] *= corr;
#pragma unroll
      for (int ot = 0; ot < 4; ++ot) oacc[ot][r] *= corr;
      float ps = 0.f;
#pragma unroll
      for (int ct = 0; ct < 4; ++ct) {
        float p = __expf(sacc[ct][r] - mx);
        ps += p;
        myP[(r + 8 * hsv) * 64 + ct * 16 + nl] = (__bf16)p;
      }
#pragma unroll
      for (int off = 8; off >= 1; off >>= 1)
        ps += __shfl_xor(ps, off, 16);
      rsum[r] += ps;
    }

    // ctx += P @ V (V tile in LDS, [d][key_local], ld=64)
#pragma unroll
    for (int ks = 0; ks < 2; ++ks) {
      v16bf pf = load_frag16(myP, 64, 0, ks * 32);
#pragma unroll
      for (int ot = 0; ot < 4; ++ot) {
        v16bf vf = load_frag16(Vt, 64, ot * 16, ks * 32);
        oacc[ot] = __builtin_amdgcn_wmma_f32_16x16x32_bf16(
            false, pf, false, vf, (short)0, oacc[ot], false, false);
      }
    }
    __syncthreads(); // done reading buf[cur] before it is refilled
  }

#pragma unroll
  for (int ot = 0; ot < 4; ++ot) {
#pragma unroll
    for (int r = 0; r < 8; ++r) {
      int s = qt * 16 + r + 8 * hsv;
      int d = ot * 16 + nl;
      out[((size_t)(b * SEQ + s) * DMODEL) + h * 64 + d] = oacc[ot][r] / rsum[r];
    }
  }
}

extern "C" void kernel_launch(void* const* d_in, const int* in_sizes, int n_in,
                              void* d_out, int out_size, void* d_ws, size_t ws_size,
                              hipStream_t stream) {
  (void)in_sizes; (void)n_in; (void)out_size; (void)ws_size;
  const float* v1   = (const float*)d_in[0];
  const float* mask = (const float*)d_in[1];
  const float* Wq   = (const float*)d_in[2];
  const float* bq   = (const float*)d_in[3];
  const float* Wk   = (const float*)d_in[4];
  const float* bk   = (const float*)d_in[5];
  const float* Wv   = (const float*)d_in[6];
  const float* bv   = (const float*)d_in[7];
  float* out = (float*)d_out;

  char* ws = (char*)d_ws;
  size_t off = 0;
  auto carve = [&](size_t bytes) -> void* {
    void* p = ws + off;
    off += (bytes + 255) & ~(size_t)255;
    return p;
  };
  __bf16* X   = (__bf16*)carve((size_t)BATCH * SEQ * DMODEL * 2);
  __bf16* WqT = (__bf16*)carve((size_t)DMODEL * DMODEL * 2);
  __bf16* WkT = (__bf16*)carve((size_t)DMODEL * DMODEL * 2);
  __bf16* WvT = (__bf16*)carve((size_t)DMODEL * DMODEL * 2);
  __bf16* Qh  = (__bf16*)carve((size_t)BATCH * SEQ * DMODEL * 2);
  __bf16* Kh  = (__bf16*)carve((size_t)BATCH * SEQ * DMODEL * 2);
  __bf16* Vt  = (__bf16*)carve((size_t)BATCH * SEQ * DMODEL * 2);

  int nX = BATCH * SEQ * DMODEL;
  cast_bf16_kernel<<<(nX + 255) / 256, 256, 0, stream>>>(v1, X, nX);
  int nW = DMODEL * DMODEL;
  transpose_cast_kernel<<<(nW + 255) / 256, 256, 0, stream>>>(Wq, WqT);
  transpose_cast_kernel<<<(nW + 255) / 256, 256, 0, stream>>>(Wk, WkT);
  transpose_cast_kernel<<<(nW + 255) / 256, 256, 0, stream>>>(Wv, WvT);

  // 256 M-tiles * 12 N64-tiles * 3 projections = 9216 waves, 8 waves/block
  qkv_gemm_kernel<<<9216 / 8, 256, 0, stream>>>(X, WqT, WkT, WvT, bq, bk, bv,
                                                Qh, Kh, Vt);
  // 24 head-batches * 16 query-blocks (8 tiles each) = 384 blocks
  attn_kernel<<<24 * 16, 256, 0, stream>>>(Qh, Kh, Vt, mask, out);
}
